// Contextualizer_57870389346593
// MI455X (gfx1250) — matrix operand order repaired and checked by reading, not compiled, not run on hardware
//
#include <hip/hip_runtime.h>
#include <hip/hip_bf16.h>

// ---------------------------------------------------------------------------
// Contextualizer odd-layer: cosine-similarity attention, sum-normalized, gated.
//   x [B=4, T=2048, D=2048] f32 -> out [B, T, 1024] f32
// Compute-bound: ~68.7 GFLOP of GEMM vs ~100 MB HBM traffic (~4us @ 23.3TB/s).
// v_wmma_f32_16x16x32_f16 (8x FLOP/instr of the f32 WMMA), f32 accumulate.
// Row normalization is linear -> folded into GEMM2 epilogue:
//   out = (S @ x0) / (rowsum(S)+eps) * x1.
// GEMM tiles are double-buffered in LDS via the CDNA5 async global->LDS engine
// (ASYNCcnt): tile s+1 streams in while WMMAs consume tile s. The pipeline
// loop is explicitly 2-step unrolled with branchless staging so the register
// allocator keeps one in-place accumulator set (no cross-copy v_movs).
// ---------------------------------------------------------------------------

typedef _Float16 half_t;
typedef __attribute__((ext_vector_type(16))) _Float16 v16h;
typedef __attribute__((ext_vector_type(8)))  _Float16 v8h;
typedef __attribute__((ext_vector_type(8)))  float    v8f;
typedef __attribute__((ext_vector_type(4)))  int      v4i;

#define BATCH 4
#define TSEQ  2048
#define DFULL 2048
#define HD    1024
#define EPSV  1e-6f
#define KPAD  40   // 32 k-halves + 8 pad halves: row stride 80B, bank-conflict-free

union U16H { v16h v; v8h h[2]; };

#if defined(__has_builtin)
#if __has_builtin(__builtin_amdgcn_global_load_async_to_lds_b128)
#define USE_ASYNC_LDS 1
#endif
#endif

__device__ __forceinline__ void cp16(const half_t* g, half_t* l) {
#ifdef USE_ASYNC_LDS
    __builtin_amdgcn_global_load_async_to_lds_b128(
        (__attribute__((address_space(1))) v4i*)(half_t*)g,
        (__attribute__((address_space(3))) v4i*)l, 0, 0);
#else
    *(v8h*)l = *(const v8h*)g;   // sync fallback: load + ds_store
#endif
}

__device__ __forceinline__ void wait_stage() {
#ifdef USE_ASYNC_LDS
#if __has_builtin(__builtin_amdgcn_s_wait_asynccnt)
    __builtin_amdgcn_s_wait_asynccnt(0);
#else
    asm volatile("s_wait_asynccnt 0" ::: "memory");
#endif
#endif
}

// Stage one 128x32 A tile and one 128x32 B tile (f16) into LDS.
// 512 16-byte chunks per tile, 256 threads -> 2 chunks each per tile.
__device__ __forceinline__ void stage_pair(const half_t* __restrict__ Ag, size_t lda,
                                           const half_t* __restrict__ Bg, size_t ldb,
                                           int kk, half_t* As, half_t* Bs, int tid) {
#pragma unroll
    for (int c = 0; c < 2; ++c) {
        const int ch = tid + c * 256;
        const int row = ch >> 2, part = ch & 3;
        cp16(Ag + (size_t)row * lda + kk + part * 8, As + row * KPAD + part * 8);
    }
#pragma unroll
    for (int c = 0; c < 2; ++c) {
        const int ch = tid + c * 256;
        const int row = ch >> 2, part = ch & 3;
        cp16(Bg + (size_t)row * ldb + kk + part * 8, Bs + row * KPAD + part * 8);
    }
}

// 8 WMMAs against one LDS buffer pair; accumulates acc[0..7] in place.
__device__ __forceinline__ void mma_step(const half_t* As, const half_t* Bs,
                                         int wave, int hh, int r, v8f acc[8]) {
    U16H ua;
    const half_t* ap = As + (wave * 16 + r) * KPAD + hh * 8;
    ua.h[0] = *(const v8h*)(ap);
    ua.h[1] = *(const v8h*)(ap + 16);
#pragma unroll
    for (int j = 0; j < 8; ++j) {
        U16H ub;
        const half_t* bp = Bs + (j * 16 + r) * KPAD + hh * 8;
        ub.h[0] = *(const v8h*)(bp);
        ub.h[1] = *(const v8h*)(bp + 16);
        acc[j] = __builtin_amdgcn_wmma_f32_16x16x32_f16(
            false, ua.v, false, ub.v, (short)0, acc[j], false, false);
    }
}

// ---------------------------------------------------------------------------
// K1: per-row sum of squares; write NORMALIZED x0 in f16 (GEMM1 operand).
// ---------------------------------------------------------------------------
__global__ __launch_bounds__(256) void k_norm(const float* __restrict__ x,
                                              half_t* __restrict__ nx0) {
    const int b = blockIdx.y, t = blockIdx.x, tid = threadIdx.x;
    const float* row = x + ((size_t)b * TSEQ + t) * DFULL;   // x0 = first HD
    float v[4];
    float ss = 0.f;
#pragma unroll
    for (int i = 0; i < 4; ++i) { v[i] = row[tid + i * 256]; ss += v[i] * v[i]; }
    __shared__ float red[256];
    red[tid] = ss;
    __syncthreads();
    for (int s = 128; s > 0; s >>= 1) {
        if (tid < s) red[tid] += red[tid + s];
        __syncthreads();
    }
    const float inv = rsqrtf(red[0] + EPSV);
    half_t* orow = nx0 + ((size_t)b * TSEQ + t) * HD;
#pragma unroll
    for (int i = 0; i < 4; ++i) orow[tid + i * 256] = (half_t)(v[i] * inv);
}

// ---------------------------------------------------------------------------
// K2: transpose raw x0 -> f16 x0t [B][HD][TSEQ] via LDS tiles so GEMM2's
// B-matrix columns (features) are contiguous in memory.
// ---------------------------------------------------------------------------
__global__ __launch_bounds__(256) void k_transpose(const float* __restrict__ x,
                                                   half_t* __restrict__ x0t) {
    const int b  = blockIdx.z;
    const int tb = blockIdx.x * 64;   // time base
    const int fb = blockIdx.y * 64;   // feature base
    __shared__ half_t tile[64][65];
    const int c  = threadIdx.x & 63;
    const int r0 = threadIdx.x >> 6;
#pragma unroll
    for (int i = 0; i < 16; ++i) {
        const int r = r0 + i * 4;
        tile[r][c] = (half_t)x[((size_t)b * TSEQ + tb + r) * DFULL + fb + c];
    }
    __syncthreads();
#pragma unroll
    for (int i = 0; i < 16; ++i) {
        const int f = r0 + i * 4;
        x0t[((size_t)b * HD + fb + f) * TSEQ + tb + c] = tile[c][f];
    }
}

// ---------------------------------------------------------------------------
// K3: GEMM1  sim = nx0 @ nx0^T  (f16 in, f32 acc, f16 out).
// Block = 8 waves; block tile 128x128; wave tile 16x128 (8 C-tiles).
// Double-buffered LDS staging, K = 1024 stepped by 32, 2-step unrolled.
// ---------------------------------------------------------------------------
__global__ __launch_bounds__(256) void k_gemm1(const half_t* __restrict__ nx0,
                                               half_t* __restrict__ simh) {
    const int b    = blockIdx.y;
    const int tm   = blockIdx.x >> 4;
    const int tn   = blockIdx.x & 15;
    const int tid  = threadIdx.x;
    const int wave = tid >> 5;
    const int lane = tid & 31;
    const int hh   = lane >> 4;
    const int r    = lane & 15;
    const int rowBase = tm * 128 + wave * 16;
    const int colBase = tn * 128;
    const half_t* Ab = nx0 + (size_t)b * TSEQ * HD;
    const half_t* Ag = Ab + (size_t)(tm * 128) * HD;
    const half_t* Bg = Ab + (size_t)colBase * HD;

    __shared__ alignas(16) half_t As[2][128][KPAD];
    __shared__ alignas(16) half_t Bs[2][128][KPAD];

    v8f acc[8] = {};
    stage_pair(Ag, HD, Bg, HD, 0, &As[0][0][0], &Bs[0][0][0], tid);

    const int NS = HD / 32;                    // 32 steps (even)
    for (int s = 0; s < NS; s += 2) {
        // step s: consume buf0, prefetch s+1 -> buf1 (branchless clamp)
        wait_stage();
        __syncthreads();
        const int k1 = (s + 1 < NS ? s + 1 : 0) * 32;
        stage_pair(Ag, HD, Bg, HD, k1, &As[1][0][0], &Bs[1][0][0], tid);
        mma_step(&As[0][0][0], &Bs[0][0][0], wave, hh, r, acc);
        // step s+1: consume buf1, prefetch s+2 -> buf0
        wait_stage();
        __syncthreads();
        const int k2 = (s + 2 < NS ? s + 2 : 0) * 32;
        stage_pair(Ag, HD, Bg, HD, k2, &As[0][0][0], &Bs[0][0][0], tid);
        mma_step(&As[1][0][0], &Bs[1][0][0], wave, hh, r, acc);
    }

    half_t* S = simh + (size_t)b * TSEQ * TSEQ;
#pragma unroll
    for (int j = 0; j < 8; ++j) {
        const int col = colBase + j * 16 + r;
#pragma unroll
        for (int i = 0; i < 8; ++i) {
            const int rr = rowBase + i + hh * 8;
            S[(size_t)rr * TSEQ + col] = (half_t)acc[j][i];
        }
    }
}

// ---------------------------------------------------------------------------
// K4: f32 row sums of sim.
// ---------------------------------------------------------------------------
__global__ __launch_bounds__(256) void k_rowsum(const half_t* __restrict__ simh,
                                                float* __restrict__ rsum) {
    const int b = blockIdx.y, t = blockIdx.x, tid = threadIdx.x;
    const half_t* row = simh + ((size_t)b * TSEQ + t) * TSEQ;
    float s = 0.f;
#pragma unroll
    for (int i = 0; i < 8; ++i) s += (float)row[tid + i * 256];
    __shared__ float red[256];
    red[tid] = s;
    __syncthreads();
    for (int st = 128; st > 0; st >>= 1) {
        if (tid < st) red[tid] += red[tid + st];
        __syncthreads();
    }
    if (tid == 0) rsum[(size_t)b * TSEQ + t] = red[0];
}

// ---------------------------------------------------------------------------
// K5: GEMM2  out = (sim @ x0) / (rowsum+eps) * x1.  A = simh, B cols from x0t.
// Same double-buffered LDS pipeline, K = 2048 stepped by 32, 2-step unrolled.
// ---------------------------------------------------------------------------
__global__ __launch_bounds__(256) void k_gemm2(const half_t* __restrict__ simh,
                                               const half_t* __restrict__ x0t,
                                               const float* __restrict__ rsum,
                                               const float* __restrict__ x,
                                               float* __restrict__ out) {
    const int b    = blockIdx.y;
    const int tm   = blockIdx.x >> 3;
    const int tn   = blockIdx.x & 7;
    const int tid  = threadIdx.x;
    const int wave = tid >> 5;
    const int lane = tid & 31;
    const int hh   = lane >> 4;
    const int r    = lane & 15;
    const int rowBase = tm * 128 + wave * 16;
    const int colBase = tn * 128;
    const half_t* Ag = simh + (size_t)b * TSEQ * TSEQ + (size_t)(tm * 128) * TSEQ;
    const half_t* Bg = x0t + (size_t)b * HD * TSEQ + (size_t)colBase * TSEQ;

    __shared__ alignas(16) half_t As[2][128][KPAD];
    __shared__ alignas(16) half_t Bs[2][128][KPAD];

    v8f acc[8] = {};
    stage_pair(Ag, TSEQ, Bg, TSEQ, 0, &As[0][0][0], &Bs[0][0][0], tid);

    const int NS = TSEQ / 32;                  // 64 steps (even)
    for (int s = 0; s < NS; s += 2) {
        wait_stage();
        __syncthreads();
        const int k1 = (s + 1 < NS ? s + 1 : 0) * 32;
        stage_pair(Ag, TSEQ, Bg, TSEQ, k1, &As[1][0][0], &Bs[1][0][0], tid);
        mma_step(&As[0][0][0], &Bs[0][0][0], wave, hh, r, acc);

        wait_stage();
        __syncthreads();
        const int k2 = (s + 2 < NS ? s + 2 : 0) * 32;
        stage_pair(Ag, TSEQ, Bg, TSEQ, k2, &As[0][0][0], &Bs[0][0][0], tid);
        mma_step(&As[1][0][0], &Bs[1][0][0], wave, hh, r, acc);
    }

#pragma unroll
    for (int i = 0; i < 8; ++i) {
        const int rr = rowBase + i + hh * 8;
        const float rinv = 1.0f / (rsum[(size_t)b * TSEQ + rr] + EPSV);
#pragma unroll
        for (int j = 0; j < 8; ++j) {
            const int col = colBase + j * 16 + r;
            const float x1v = x[((size_t)b * TSEQ + rr) * DFULL + HD + col];
            out[((size_t)b * TSEQ + rr) * HD + col] = acc[j][i] * rinv * x1v;
        }
    }
}

// ---------------------------------------------------------------------------
// Launch. Workspace layout (bytes):
//   [0, 16M)    nx0  f16 [B][T][HD]   normalized x0
//   [16M, 32M)  x0t  f16 [B][HD][T]   raw x0 transposed
//   [32M, 64M)  simh f16 [B][T][T]    cosine-sim matrix
//   [64M, +32K) rsum f32 [B][T]
// ---------------------------------------------------------------------------
extern "C" void kernel_launch(void* const* d_in, const int* in_sizes, int n_in,
                              void* d_out, int out_size, void* d_ws, size_t ws_size,
                              hipStream_t stream) {
    const float* x = (const float*)d_in[0];
    float* out = (float*)d_out;
    char* ws = (char*)d_ws;

    const size_t NX0_B = (size_t)BATCH * TSEQ * HD * sizeof(half_t);   // 16 MB
    const size_t X0T_B = (size_t)BATCH * HD * TSEQ * sizeof(half_t);   // 16 MB
    const size_t SIM_B = (size_t)BATCH * TSEQ * TSEQ * sizeof(half_t); // 32 MB

    half_t* nx0  = (half_t*)(ws);
    half_t* x0t  = (half_t*)(ws + NX0_B);
    half_t* simh = (half_t*)(ws + NX0_B + X0T_B);
    float*  rsum = (float*)(ws + NX0_B + X0T_B + SIM_B);

    k_norm<<<dim3(TSEQ, BATCH), 256, 0, stream>>>(x, nx0);
    k_transpose<<<dim3(TSEQ / 64, HD / 64, BATCH), 256, 0, stream>>>(x, x0t);
    k_gemm1<<<dim3(16 * 16, BATCH), 256, 0, stream>>>(nx0, simh);
    k_rowsum<<<dim3(TSEQ, BATCH), 256, 0, stream>>>(simh, rsum);
    k_gemm2<<<dim3(16 * 8, BATCH), 256, 0, stream>>>(simh, x0t, rsum, x, out);
}
